// Attention_19911468384972
// MI455X (gfx1250) — matrix operand order repaired and checked by reading, not compiled, not run on hardware
//
#include <hip/hip_runtime.h>

// ---------------------------------------------------------------------------
// Causal MHA (b=2, n=2048, dim=1024, h=16, d=64) for MI455X / gfx1250.
// All matmuls via v_wmma_f32_16x16x32_f16 (f16 in, f32 accumulate).
// GEMM staging uses GLOBAL_LOAD_ASYNC_TO_LDS_B128 + double-buffered LDS so
// global latency hides behind the WMMA stream (ASYNCcnt-tracked DMA, no VGPR
// round-trip).
// ---------------------------------------------------------------------------

typedef _Float16 half_t;
typedef __attribute__((ext_vector_type(16))) _Float16 v16h;
typedef __attribute__((ext_vector_type(8)))  _Float16 v8h;
typedef __attribute__((ext_vector_type(8)))  float    v8f;

#define WMMA16(a, b, c) \
    __builtin_amdgcn_wmma_f32_16x16x32_f16(false, (a), false, (b), (short)0, (c), false, false)

static __device__ __forceinline__ v8f zero8() {
    v8f z = {0.f, 0.f, 0.f, 0.f, 0.f, 0.f, 0.f, 0.f};
    return z;
}

// Async cache->LDS 16-byte copy (per-lane addresses). lds_off = addr[31:0] of
// the shared-memory destination; gp = global source. Tracked by ASYNCcnt.
static __device__ __forceinline__ void async_ld_b128(unsigned lds_off, const half_t* gp) {
    asm volatile("global_load_async_to_lds_b128 %0, %1, off"
                 :: "v"(lds_off), "v"(gp) : "memory");
}
static __device__ __forceinline__ void wait_async0() {
    asm volatile("s_wait_asynccnt 0x0" ::: "memory");
}

// A operand: 16x32 f16 tile, row-major source with leading dim `ld` (elements).
// ISA layout: lanes 0-15 -> M=0..15, halves {0..7}=K0-7, {8..15}=K16-23;
//             lanes 16-31 -> same M, halves {0..7}=K8-15, {8..15}=K24-31.
static __device__ __forceinline__ v16h load_a_frag(const half_t* __restrict__ base, int ld) {
    int lane = threadIdx.x & 31;
    int m  = lane & 15;
    int kh = (lane >> 4) << 3;            // 0 or 8
    const half_t* p = base + (size_t)m * ld;
    v8h lo = *(const v8h*)(p + kh);       // K = kh .. kh+7
    v8h hi = *(const v8h*)(p + 16 + kh);  // K = 16+kh .. 16+kh+7
    return __builtin_shufflevector(lo, hi, 0,1,2,3,4,5,6,7,8,9,10,11,12,13,14,15);
}

// B operand: 32x16 f16 (KxN), sourced from B^T stored row-major [N][K], ld in elems.
// ISA layout: lanes 0-15 -> N=0..15, halves {0..7}=K0-7, {8..15}=K16-23; lanes 16-31 -> K8-15, K24-31.
static __device__ __forceinline__ v16h load_bt_frag(const half_t* __restrict__ base, int ld) {
    int lane = threadIdx.x & 31;
    int nn = lane & 15;
    int kk = (lane >> 4) << 3;            // 0 or 8
    const half_t* p = base + (size_t)nn * ld + kk;
    v8h lo = *(const v8h*)(p);
    v8h hi = *(const v8h*)(p + 16);
    return __builtin_shufflevector(lo, hi, 0,1,2,3,4,5,6,7,8,9,10,11,12,13,14,15);
}

// ------------------------------- converters --------------------------------

__global__ void cvt_f32_f16(const float* __restrict__ s, half_t* __restrict__ d, int n) {
    int i = blockIdx.x * blockDim.x + threadIdx.x;
    if (i < n) d[i] = (half_t)s[i];
}

// src: [rows][cols] f32 -> dst: [cols][rows] f16  (weight transpose for B-operand)
__global__ void cvt_transpose_f16(const float* __restrict__ s, half_t* __restrict__ d,
                                  int rows, int cols) {
    int i = blockIdx.x * blockDim.x + threadIdx.x;
    if (i < rows * cols) {
        int r = i / cols, c = i % cols;
        d[(size_t)c * rows + r] = (half_t)s[i];
    }
}

// --------------------------------- GEMM ------------------------------------
// C[M][N] = A[M][K] @ Bt[N][K]^T + bias[N]
// mode 0: f16 row-major store; mode 1: f16 store transposed to vT[b,h][d][n];
// mode 2: f32 row-major store.
__global__ __launch_bounds__(256) void gemm_wmma(
    const half_t* __restrict__ A, const half_t* __restrict__ Bt,
    const float* __restrict__ bias, half_t* __restrict__ Ch,
    float* __restrict__ Cf, int M, int N, int K, int mode) {

    __shared__ half_t As[2][128 * 40];
    __shared__ half_t Bs[2][128 * 40];

    const int n0 = blockIdx.x * 128;
    const int m0 = blockIdx.y * 128;
    const int tid  = threadIdx.x;
    const int wid  = tid >> 5;
    const int lane = tid & 31;
    const int wm = (wid & 3) * 32;   // 4 wave-rows of 32
    const int wn = (wid >> 2) * 64;  // 2 wave-cols of 64

    // Staging: 128x32 f16 tile = 512 x 16B chunks per matrix; 2 per thread.
    // Thread's chunks: (row r0, col cc0) and (row r0+64, col cc0).
    const int r0  = tid >> 2;
    const int cc0 = (tid & 3) << 3;
    const half_t* ga0 = A  + (size_t)(m0 + r0)      * K + cc0;
    const half_t* ga1 = A  + (size_t)(m0 + r0 + 64) * K + cc0;
    const half_t* gb0 = Bt + (size_t)(n0 + r0)      * K + cc0;
    const half_t* gb1 = Bt + (size_t)(n0 + r0 + 64) * K + cc0;
    unsigned lA[2], lA64[2], lB[2], lB64[2];
#pragma unroll
    for (int bf = 0; bf < 2; bf++) {
        lA[bf]   = (unsigned)(uintptr_t)&As[bf][r0 * 40 + cc0];
        lA64[bf] = (unsigned)(uintptr_t)&As[bf][(r0 + 64) * 40 + cc0];
        lB[bf]   = (unsigned)(uintptr_t)&Bs[bf][r0 * 40 + cc0];
        lB64[bf] = (unsigned)(uintptr_t)&Bs[bf][(r0 + 64) * 40 + cc0];
    }

    auto stage = [&](int bf, int koff) {
        async_ld_b128(lA[bf],   ga0 + koff);
        async_ld_b128(lA64[bf], ga1 + koff);
        async_ld_b128(lB[bf],   gb0 + koff);
        async_ld_b128(lB64[bf], gb1 + koff);
    };

    v8f acc[2][4];
#pragma unroll
    for (int i = 0; i < 2; i++)
#pragma unroll
        for (int t = 0; t < 4; t++) acc[i][t] = zero8();

    // prologue: fill buffer 0
    stage(0, 0);
    wait_async0();
    __syncthreads();

    int cur = 0;
    for (int kb = 0; kb < K; kb += 32) {
        // prefetch next K-chunk into the other buffer while we compute
        if (kb + 32 < K) stage(cur ^ 1, kb + 32);

        v16h af0 = load_a_frag(&As[cur][(wm + 0)  * 40], 40);
        v16h af1 = load_a_frag(&As[cur][(wm + 16) * 40], 40);
#pragma unroll
        for (int t = 0; t < 4; t++) {
            v16h bfr = load_bt_frag(&Bs[cur][(wn + t * 16) * 40], 40);
            acc[0][t] = WMMA16(af0, bfr, acc[0][t]);
            acc[1][t] = WMMA16(af1, bfr, acc[1][t]);
        }

        wait_async0();       // next buffer's DMA complete
        __syncthreads();     // everyone done reading cur / data visible
        cur ^= 1;
    }

    const int ln15 = lane & 15;
    const int hi8  = (lane >> 4) << 3;
#pragma unroll
    for (int im = 0; im < 2; im++) {
#pragma unroll
        for (int t = 0; t < 4; t++) {
            int col = n0 + wn + t * 16 + ln15;
            float bv = bias[col];
#pragma unroll
            for (int v = 0; v < 8; v++) {
                int row = m0 + wm + im * 16 + v + hi8;
                float val = acc[im][t][v] + bv;
                if (mode == 0) {
                    Ch[(size_t)row * N + col] = (half_t)val;
                } else if (mode == 1) {
                    // col = h*64 + dd ; row = b*2048 + i  ->  vT[((b*16+h)*64+dd)*2048 + i]
                    int hh = col >> 6, dd = col & 63;
                    int bb = row >> 11, ii = row & 2047;
                    Ch[(((size_t)bb * 16 + hh) * 64 + dd) * 2048 + ii] = (half_t)val;
                } else {
                    Cf[(size_t)row * N + col] = val;
                }
            }
        }
    }
}

// ------------------------------ attention ----------------------------------
// Flash-style causal attention. Grid: (n/128, b*h). Each wave owns 16 query
// rows; loops over 64-key blocks up to the diagonal with online softmax.
__global__ __launch_bounds__(256) void attn_fwd(
    const half_t* __restrict__ q, const half_t* __restrict__ k,
    const half_t* __restrict__ vT, half_t* __restrict__ o) {

    constexpr int N = 2048, DIM = 1024, D = 64;
    constexpr float SCALE = 0.125f;        // 64^-0.5
    constexpr float MASK_VAL = -50000.0f;
    constexpr int PLD = 72;                // P-tile LDS leading dim (halves)

    __shared__ half_t Ptile[8 * 16 * PLD]; // wave-private 16x64 P staging

    const int bh   = blockIdx.y;           // b*16 + h
    const int b    = bh >> 4;
    const int h    = bh & 15;
    const int wid  = threadIdx.x >> 5;
    const int lane = threadIdx.x & 31;
    const int i0   = blockIdx.x * 128 + wid * 16;  // this wave's first query row

    const int ln15 = lane & 15;
    const int hi8  = (lane >> 4) << 3;

    const half_t* qbase = q + ((size_t)b * N + i0) * DIM + h * D;
    const half_t* kbase = k + ((size_t)b * N) * DIM + h * D;
    const half_t* vbase = vT + ((size_t)bh * D) * N;
    half_t* Pw = &Ptile[wid * 16 * PLD];

    // Q fragments for K-dim chunks d=0..31 and d=32..63 (held for whole loop)
    v16h qf0 = load_a_frag(qbase + 0,  DIM);
    v16h qf1 = load_a_frag(qbase + 32, DIM);

    v8f oacc[4];
#pragma unroll
    for (int t = 0; t < 4; t++) oacc[t] = zero8();
    float mstat[8], lstat[8];
#pragma unroll
    for (int v = 0; v < 8; v++) { mstat[v] = -3.0e38f; lstat[v] = 0.f; }

    for (int j0 = 0; j0 < i0 + 16; j0 += 64) {
        // ---- S = (Q @ K^T) over 64 keys: 4 N-tiles x 2 K-steps ----
        v8f s[4];
#pragma unroll
        for (int t = 0; t < 4; t++) {
            v8f sa = zero8();
            const half_t* kb0 = kbase + (size_t)(j0 + t * 16) * DIM;
            v16h bf0 = load_bt_frag(kb0 + 0,  DIM);
            v16h bf1 = load_bt_frag(kb0 + 32, DIM);
            sa = WMMA16(qf0, bf0, sa);
            sa = WMMA16(qf1, bf1, sa);
            s[t] = sa;
        }

        // ---- scale + causal mask ----
#pragma unroll
        for (int t = 0; t < 4; t++) {
#pragma unroll
            for (int v = 0; v < 8; v++) {
                int i = i0 + v + hi8;
                int j = j0 + t * 16 + ln15;
                float val = s[t][v] * SCALE;
                if (j > i) val = MASK_VAL;
                s[t][v] = val;
            }
        }

        // ---- online softmax (row stats across 16-lane groups) ----
#pragma unroll
        for (int v = 0; v < 8; v++) {
            float pm = -3.0e38f;
#pragma unroll
            for (int t = 0; t < 4; t++) pm = fmaxf(pm, s[t][v]);
#pragma unroll
            for (int off = 1; off < 16; off <<= 1)
                pm = fmaxf(pm, __shfl_xor(pm, off, 32));
            float newm = fmaxf(mstat[v], pm);
            float corr = __expf(mstat[v] - newm);
            mstat[v] = newm;
            float rs = 0.f;
#pragma unroll
            for (int t = 0; t < 4; t++) {
                float e = __expf(s[t][v] - newm);
                s[t][v] = e;
                rs += e;
            }
#pragma unroll
            for (int off = 1; off < 16; off <<= 1)
                rs += __shfl_xor(rs, off, 32);
            lstat[v] = lstat[v] * corr + rs;
#pragma unroll
            for (int t = 0; t < 4; t++) oacc[t][v] *= corr;
        }

        // ---- stage P (16x64 f16) into wave-private LDS (re-layout D -> A) ----
#pragma unroll
        for (int t = 0; t < 4; t++)
#pragma unroll
            for (int v = 0; v < 8; v++)
                Pw[(v + hi8) * PLD + t * 16 + ln15] = (half_t)s[t][v];

        // ---- O += P @ V : 4 dd-tiles x 2 K-steps over 64 keys ----
#pragma unroll
        for (int jk = 0; jk < 2; jk++) {
            v16h af = load_a_frag(Pw + jk * 32, PLD);
#pragma unroll
            for (int t = 0; t < 4; t++) {
                v16h bf = load_bt_frag(vbase + (size_t)(t * 16) * N + j0 + jk * 32, N);
                oacc[t] = WMMA16(af, bf, oacc[t]);
            }
        }
    }

    // ---- normalize and store O (f16, [b][i][h*64+dd]) ----
#pragma unroll
    for (int t = 0; t < 4; t++) {
#pragma unroll
        for (int v = 0; v < 8; v++) {
            int i  = i0 + v + hi8;
            int dd = t * 16 + ln15;
            float val = oacc[t][v] / lstat[v];
            o[((size_t)b * N + i) * DIM + h * D + dd] = (half_t)val;
        }
    }
}

// ------------------------------- launcher ----------------------------------

extern "C" void kernel_launch(void* const* d_in, const int* in_sizes, int n_in,
                              void* d_out, int out_size, void* d_ws, size_t ws_size,
                              hipStream_t stream) {
    const float* x   = (const float*)d_in[0];
    const float* Wq  = (const float*)d_in[1];
    const float* bq  = (const float*)d_in[2];
    const float* Wkv = (const float*)d_in[3];
    const float* bkv = (const float*)d_in[4];
    const float* Wo  = (const float*)d_in[5];
    const float* bo  = (const float*)d_in[6];

    // workspace layout (f16 elements)
    half_t* ws   = (half_t*)d_ws;
    half_t* xh   = ws;                        // 4M   : x as f16          (b*n, dim)
    half_t* wqT  = xh   + 4194304;            // 1M   : Wq^T   [1024][1024]
    half_t* wkvT = wqT  + 1048576;            // 2M   : Wkv^T  [2048][1024]
    half_t* woT  = wkvT + 2097152;            // 1M   : Wo^T   [1024][1024]
    half_t* qb   = woT  + 1048576;            // 4M   : Q (b*n, dim)
    half_t* kb   = qb   + 4194304;            // 4M   : K (b*n, dim)
    half_t* vt   = kb   + 4194304;            // 4M   : V^T [b,h][64][2048]
    half_t* ob   = vt   + 4194304;            // 4M   : attn out (b*n, dim)

    // 1) precision conversion / weight transposes
    cvt_f32_f16<<<4194304 / 256, 256, 0, stream>>>(x, xh, 4194304);
    cvt_transpose_f16<<<1048576 / 256, 256, 0, stream>>>(Wq,  wqT,  1024, 1024);
    cvt_transpose_f16<<<2097152 / 256, 256, 0, stream>>>(Wkv, wkvT, 1024, 2048);
    cvt_transpose_f16<<<1048576 / 256, 256, 0, stream>>>(Wo,  woT,  1024, 1024);

    // 2) QKV projections (M=4096, N=1024, K=1024)
    dim3 gg(1024 / 128, 4096 / 128);
    gemm_wmma<<<gg, 256, 0, stream>>>(xh, wqT,                 bq,         qb, nullptr,
                                      4096, 1024, 1024, 0);
    gemm_wmma<<<gg, 256, 0, stream>>>(xh, wkvT,                bkv,        kb, nullptr,
                                      4096, 1024, 1024, 0);
    gemm_wmma<<<gg, 256, 0, stream>>>(xh, wkvT + 1024 * 1024,  bkv + 1024, vt, nullptr,
                                      4096, 1024, 1024, 1);

    // 3) causal flash attention
    attn_fwd<<<dim3(2048 / 128, 32), 256, 0, stream>>>(qb, kb, vt, ob);

    // 4) output projection -> f32 d_out
    gemm_wmma<<<gg, 256, 0, stream>>>(ob, woT, bo, nullptr, (float*)d_out,
                                      4096, 1024, 1024, 2);
}
